// RNNEncoder_28527172780217
// MI455X (gfx1250) — matrix operand order repaired and checked by reading, not compiled, not run on hardware
//
#include <hip/hip_runtime.h>
#include <math.h>

#define B_ 16
#define S_ 1024
#define D_ 512
#define G3 (3 * D_)      // 1536
#define SB (S_ * B_)     // 16384
#define NWG_SCAN 16

typedef __attribute__((ext_vector_type(16))) _Float16 v16h;
typedef __attribute__((ext_vector_type(8)))  float    v8f;
typedef __attribute__((ext_vector_type(4)))  unsigned int u32x4;

union FragU { v16h v; u32x4 q[2]; };

// A-operand fragment (16x32 f16, row-major [M, ldk]):
// ISA 7.12.2: lanes 0-15 hold K = {k..k+7, k+16..k+23} of row M=lane,
// lanes 16-31 hold K = {k+8..k+15, k+24..k+31}  ->  two 16B loads 32B apart.
__device__ __forceinline__ v16h load_frag_a(const _Float16* base, int ldk, int row, int k, int hi) {
    FragU f;
    const u32x4* p = reinterpret_cast<const u32x4*>(base + (size_t)row * ldk + k + hi * 8);
    f.q[0] = p[0];
    f.q[1] = p[2];   // +16 halfs
    return f.v;
}

// B-operand fragment (32x16 f16). Column N stored as a row of W [N, ldk]:
// lanes 0-15 hold K = {k..k+7, k+16..k+23} of column N=lane, lanes 16-31 hold K = {k+8..k+15, k+24..k+31}
// -> two 16B loads 32B apart at col*ldk + k + hi*8.
__device__ __forceinline__ v16h load_frag_b(const _Float16* base, int ldk, int col, int k, int hi) {
    FragU f;
    const u32x4* p = reinterpret_cast<const u32x4*>(base + (size_t)col * ldk + k + hi * 8);
    f.q[0] = p[0];
    f.q[1] = p[2];
    return f.v;
}

__device__ __forceinline__ float sigmoidf_(float x) { return 1.0f / (1.0f + __expf(-x)); }

// ---------------------------------------------------------------- converts
__global__ void cvt_kernel(const float* __restrict__ a, _Float16* __restrict__ o, int n) {
    int i = blockIdx.x * 256 + threadIdx.x;
    if (i < n) o[i] = (_Float16)a[i];
}

// a[rows][cols] (f32) -> o[cols][rows] (f16)
__global__ void transpose_cvt_kernel(const float* __restrict__ a, _Float16* __restrict__ o,
                                     int rows, int cols) {
    int i = blockIdx.x * 256 + threadIdx.x;
    if (i < rows * cols) {
        int r = i / cols, c = i % cols;
        o[(size_t)c * rows + r] = (_Float16)a[i];
    }
}

// ---------------------------------------------------------------- layernorms
// LN over inputs [B,S,D] -> time-major f16 [S,B,D]
__global__ __launch_bounds__(256) void ln0_kernel(const float* __restrict__ in,
                                                  const float* __restrict__ g,
                                                  const float* __restrict__ b,
                                                  _Float16* __restrict__ x16) {
    __shared__ float sm[16];
    const int r = blockIdx.x;              // output row = s*B + b
    const int s = r / B_, bb = r % B_;
    const int t = threadIdx.x;
    const float* src = in + ((size_t)bb * S_ + s) * D_;
    float v0 = src[t], v1 = src[t + 256];
    float sum = v0 + v1, sq = v0 * v0 + v1 * v1;
    for (int off = 16; off; off >>= 1) {
        sum += __shfl_xor(sum, off, 32);
        sq  += __shfl_xor(sq,  off, 32);
    }
    const int wid = t >> 5, lid = t & 31;
    if (lid == 0) { sm[wid] = sum; sm[8 + wid] = sq; }
    __syncthreads();
    if (t == 0) {
        float a = 0.f, c = 0.f;
        for (int i = 0; i < 8; ++i) { a += sm[i]; c += sm[8 + i]; }
        sm[0] = a; sm[8] = c;
    }
    __syncthreads();
    const float mean = sm[0] * (1.0f / D_);
    const float var  = sm[8] * (1.0f / D_) - mean * mean;
    const float rstd = rsqrtf(var + 1e-5f);
    _Float16* dst = x16 + (size_t)r * D_;
    dst[t]       = (_Float16)((v0 - mean) * rstd * g[t] + b[t]);
    dst[t + 256] = (_Float16)((v1 - mean) * rstd * g[t + 256] + b[t + 256]);
}

// out = inputs + h^T (residual 1); y16 = f16(LN(out))
__global__ __launch_bounds__(256) void res_ln1_kernel(const float* __restrict__ in,
                                                      const float* __restrict__ xf,
                                                      const float* __restrict__ g,
                                                      const float* __restrict__ b,
                                                      float* __restrict__ outres,
                                                      _Float16* __restrict__ y16) {
    __shared__ float sm[16];
    const int r = blockIdx.x;              // row = b*S + s
    const int bb = r / S_, s = r % S_;
    const int t = threadIdx.x;
    const float* xr = xf + ((size_t)s * B_ + bb) * D_;
    const float* ir = in + (size_t)r * D_;
    float v0 = ir[t] + xr[t], v1 = ir[t + 256] + xr[t + 256];
    outres[(size_t)r * D_ + t] = v0;
    outres[(size_t)r * D_ + t + 256] = v1;
    float sum = v0 + v1, sq = v0 * v0 + v1 * v1;
    for (int off = 16; off; off >>= 1) {
        sum += __shfl_xor(sum, off, 32);
        sq  += __shfl_xor(sq,  off, 32);
    }
    const int wid = t >> 5, lid = t & 31;
    if (lid == 0) { sm[wid] = sum; sm[8 + wid] = sq; }
    __syncthreads();
    if (t == 0) {
        float a = 0.f, c = 0.f;
        for (int i = 0; i < 8; ++i) { a += sm[i]; c += sm[8 + i]; }
        sm[0] = a; sm[8] = c;
    }
    __syncthreads();
    const float mean = sm[0] * (1.0f / D_);
    const float var  = sm[8] * (1.0f / D_) - mean * mean;
    const float rstd = rsqrtf(var + 1e-5f);
    _Float16* dst = y16 + (size_t)r * D_;
    dst[t]       = (_Float16)((v0 - mean) * rstd * g[t] + b[t]);
    dst[t + 256] = (_Float16)((v1 - mean) * rstd * g[t + 256] + b[t + 256]);
}

// ---------------------------------------------------------------- generic WMMA GEMM
// C[M,N] = A[M,K](f16) * W[N,K]^T(f16) + bias, with epilogue:
//   mode 0: outF = v                (gi projection)
//   mode 1: outH = f16(gelu(v))     (FFN layer 1)
//   mode 2: outF = v + res          (FFN layer 2 + residual 2)
// Block = 256 threads = 8 waves (4 M x 2 N); wave tile 16x64; block tile 64x128.
__global__ __launch_bounds__(256) void gemm_f16_kernel(const _Float16* __restrict__ A,
                                                       const _Float16* __restrict__ W,
                                                       const float* __restrict__ bias,
                                                       const float* __restrict__ res,
                                                       float* __restrict__ outF,
                                                       _Float16* __restrict__ outH,
                                                       int N, int K, int mode) {
    const int lane = threadIdx.x & 31;
    const int l16 = lane & 15, hi = lane >> 4;
    const int wave = threadIdx.x >> 5;
    const int mbase = blockIdx.x * 64 + (wave >> 1) * 16;
    const int nbase = blockIdx.y * 128 + (wave & 1) * 64;

    v8f acc[4] = {};
    for (int k = 0; k < K; k += 32) {
        v16h a = load_frag_a(A, K, mbase + l16, k, hi);
        v16h bf[4];
#pragma unroll
        for (int t = 0; t < 4; ++t)
            bf[t] = load_frag_b(W, K, nbase + t * 16 + l16, k, hi);
#pragma unroll
        for (int t = 0; t < 4; ++t)
            acc[t] = __builtin_amdgcn_wmma_f32_16x16x32_f16(
                false, a, false, bf[t], (short)0, acc[t], false, false);
    }
#pragma unroll
    for (int t = 0; t < 4; ++t) {
        const int n = nbase + t * 16 + l16;
        const float bn = bias[n];
#pragma unroll
        for (int j = 0; j < 8; ++j) {
            const int m = mbase + j + 8 * hi;     // C layout: VGPR j <-> M = j + 8*hi
            const size_t idx = (size_t)m * N + n;
            float v = acc[t][j] + bn;
            if (mode == 0) {
                outF[idx] = v;
            } else if (mode == 1) {
                outH[idx] = (_Float16)(0.5f * v * (1.0f + erff(v * 0.70710678f)));
            } else {
                outF[idx] = v + res[idx];
            }
        }
    }
}

// ---------------------------------------------------------------- grid barrier
__device__ __forceinline__ void grid_barrier(unsigned* ctr) {
    __threadfence();                 // release h stores to agent scope
    __syncthreads();
    if (threadIdx.x == 0) {
        __hip_atomic_fetch_add(ctr, 1u, __ATOMIC_RELEASE, __HIP_MEMORY_SCOPE_AGENT);
        unsigned v;
        do {
            v = __hip_atomic_load(ctr, __ATOMIC_ACQUIRE, __HIP_MEMORY_SCOPE_AGENT);
            if (v < NWG_SCAN) __builtin_amdgcn_s_sleep(2);
        } while (v < NWG_SCAN);
    }
    __syncthreads();
    __threadfence();                 // acquire: drop stale cached h
}

// ---------------------------------------------------------------- GRU recurrence
// 16 blocks x 2 waves; wave gw owns hidden columns [16*gw, 16*gw+16).
// Its (r,z,n) weight columns (3*16 rows of w_hh, 96KB/block) stay LDS-resident.
// h: own slice in registers; full h broadcast via ping-pong f16 buffer + spin barrier.
__global__ __launch_bounds__(64) void gru_scan_kernel(const _Float16* __restrict__ whh,
                                                      const float* __restrict__ gi,
                                                      const float* __restrict__ h0,
                                                      const float* __restrict__ bhh,
                                                      _Float16* __restrict__ xh16,
                                                      float* __restrict__ xf,
                                                      _Float16* __restrict__ hbuf,
                                                      unsigned* __restrict__ bar) {
    extern __shared__ _Float16 lds[];            // [2 waves][3 gates][16 cols][512 k]
    const int lane = threadIdx.x & 31;
    const int l16 = lane & 15, hi = lane >> 4;
    const int wv = threadIdx.x >> 5;             // wave in block: 0..1
    const int col0 = (blockIdx.x * 2 + wv) * 16; // owned hidden columns

    // Stage this wave's w_hh columns into LDS (once per layer).
    for (int t = 0; t < 48; ++t) {
        const int region = t >> 4, c = t & 15;
        const u32x4* src = reinterpret_cast<const u32x4*>(
            whh + (size_t)(region * D_ + col0 + c) * D_);
        u32x4* dst = reinterpret_cast<u32x4*>(
            lds + (size_t)((wv * 3 + region) * 16 + c) * D_);
        dst[lane]      = src[lane];
        dst[lane + 32] = src[lane + 32];
    }
    __syncthreads();

    const int n = col0 + l16;
    float hprev[8];
#pragma unroll
    for (int j = 0; j < 8; ++j) {
        const int m = j + 8 * hi;
        const float h = h0[(size_t)m * D_ + n];
        hprev[j] = h;
        hbuf[(size_t)m * D_ + n] = (_Float16)h;  // parity-0 buffer
    }
    const float br = bhh[n], bz = bhh[D_ + n], bn = bhh[2 * D_ + n];
    grid_barrier(bar);

    for (int s = 0; s < S_; ++s) {
        const _Float16* hsrc = hbuf + (size_t)(s & 1) * (B_ * D_);
        _Float16*       hdst = hbuf + (size_t)((s + 1) & 1) * (B_ * D_);

        v8f acc[3] = {};
        for (int k = 0; k < D_; k += 32) {
            v16h a = load_frag_a(hsrc, D_, l16, k, hi);        // A = h_prev (16xK)
#pragma unroll
            for (int g = 0; g < 3; ++g) {
                v16h bfr = load_frag_b(lds, D_, (wv * 3 + g) * 16 + l16, k, hi);
                acc[g] = __builtin_amdgcn_wmma_f32_16x16x32_f16(
                    false, a, false, bfr, (short)0, acc[g], false, false);
            }
        }

        const float* git = gi + (size_t)s * (B_ * G3);
        if (s + 1 < S_) __builtin_prefetch(git + (size_t)(B_ * G3), 0, 1);
#pragma unroll
        for (int j = 0; j < 8; ++j) {
            const int m = j + 8 * hi;
            const float* row = git + (size_t)m * G3;
            const float r  = sigmoidf_(row[n]          + acc[0][j] + br);
            const float z  = sigmoidf_(row[D_ + n]     + acc[1][j] + bz);
            const float nn = tanhf(row[2 * D_ + n] + r * (acc[2][j] + bn));
            const float h  = (1.0f - z) * nn + z * hprev[j];
            hprev[j] = h;
            const size_t oidx = ((size_t)s * B_ + m) * D_ + n;
            xf[oidx]   = h;
            xh16[oidx] = (_Float16)h;
            hdst[(size_t)m * D_ + n] = (_Float16)h;
        }
        grid_barrier(bar + 1 + s);
    }
}

// ---------------------------------------------------------------- host
extern "C" void kernel_launch(void* const* d_in, const int* in_sizes, int n_in,
                              void* d_out, int out_size, void* d_ws, size_t ws_size,
                              hipStream_t stream) {
    (void)in_sizes; (void)n_in; (void)out_size; (void)ws_size;
    const float* inputs = (const float*)d_in[0];   // [B,S,D]
    const float* h0     = (const float*)d_in[1];   // [3,B,D]
    const float* w_ih   = (const float*)d_in[2];   // [3,3D,D]
    const float* w_hh   = (const float*)d_in[3];   // [3,3D,D]
    const float* b_ih   = (const float*)d_in[4];   // [3,3D]
    const float* b_hh   = (const float*)d_in[5];   // [3,3D]
    const float* ln0g   = (const float*)d_in[6];
    const float* ln0b   = (const float*)d_in[7];
    const float* ln1g   = (const float*)d_in[8];
    const float* ln1b   = (const float*)d_in[9];
    const float* w1     = (const float*)d_in[10];  // [D,4D]
    const float* b1     = (const float*)d_in[11];
    const float* w2     = (const float*)d_in[12];  // [4D,D]
    const float* b2     = (const float*)d_in[13];
    float* out = (float*)d_out;

    char* wsp = (char*)d_ws;
    auto alloc = [&](size_t bytes) {
        char* p = wsp;
        wsp += (bytes + 255) & ~(size_t)255;
        return p;
    };
    _Float16* xh16   = (_Float16*)alloc((size_t)SB * D_ * 2);       // layer input, f16
    float*    gi     = (float*)   alloc((size_t)SB * G3 * 4);       // input projections
    float*    xf     = (float*)   alloc((size_t)SB * D_ * 4);       // last layer f32 out
    _Float16* wih16  = (_Float16*)alloc((size_t)3 * G3 * D_ * 2);
    _Float16* whh16  = (_Float16*)alloc((size_t)3 * G3 * D_ * 2);
    _Float16* wt1    = (_Float16*)alloc((size_t)4 * D_ * D_ * 2);   // w1^T  [2048,512]
    _Float16* wt2    = (_Float16*)alloc((size_t)4 * D_ * D_ * 2);   // w2^T  [512,2048]
    _Float16* hbuf   = (_Float16*)alloc((size_t)2 * B_ * D_ * 2);   // ping-pong h
    float*    outres = (float*)   alloc((size_t)SB * D_ * 4);       // residual-1 out
    _Float16* y16    = (_Float16*)alloc((size_t)SB * D_ * 2);       // LN1 out, f16
    _Float16* mid16  = (_Float16*)alloc((size_t)SB * 4 * D_ * 2);   // gelu(ffn1), f16
    unsigned* bars   = (unsigned*)alloc((size_t)3 * (S_ + 1) * sizeof(unsigned));

    hipMemsetAsync(bars, 0, (size_t)3 * (S_ + 1) * sizeof(unsigned), stream);
    (void)hipFuncSetAttribute((const void*)gru_scan_kernel,
                              hipFuncAttributeMaxDynamicSharedMemorySize, 96 * 1024);

    const int nw = 3 * G3 * D_;
    cvt_kernel<<<(nw + 255) / 256, 256, 0, stream>>>(w_ih, wih16, nw);
    cvt_kernel<<<(nw + 255) / 256, 256, 0, stream>>>(w_hh, whh16, nw);
    const int nf = D_ * 4 * D_;
    transpose_cvt_kernel<<<(nf + 255) / 256, 256, 0, stream>>>(w1, wt1, D_, 4 * D_);
    transpose_cvt_kernel<<<(nf + 255) / 256, 256, 0, stream>>>(w2, wt2, 4 * D_, D_);

    ln0_kernel<<<SB, 256, 0, stream>>>(inputs, ln0g, ln0b, xh16);

    for (int l = 0; l < 3; ++l) {
        gemm_f16_kernel<<<dim3(SB / 64, G3 / 128), 256, 0, stream>>>(
            xh16, wih16 + (size_t)l * G3 * D_, b_ih + (size_t)l * G3,
            nullptr, gi, nullptr, G3, D_, /*mode=*/0);
        gru_scan_kernel<<<NWG_SCAN, 64, 96 * 1024, stream>>>(
            whh16 + (size_t)l * G3 * D_, gi, h0 + (size_t)l * B_ * D_,
            b_hh + (size_t)l * G3, xh16, xf, hbuf, bars + (size_t)l * (S_ + 1));
    }

    res_ln1_kernel<<<SB, 256, 0, stream>>>(inputs, xf, ln1g, ln1b, outres, y16);

    gemm_f16_kernel<<<dim3(SB / 64, (4 * D_) / 128), 256, 0, stream>>>(
        y16, wt1, b1, nullptr, nullptr, mid16, 4 * D_, D_, /*mode=*/1);
    gemm_f16_kernel<<<dim3(SB / 64, D_ / 128), 256, 0, stream>>>(
        mid16, wt2, b2, outres, out, nullptr, D_, 4 * D_, /*mode=*/2);
}
